// AttnBlock2D_22136261444337
// MI455X (gfx1250) — matrix operand order repaired and checked by reading, not compiled, not run on hardware
//
#include <hip/hip_runtime.h>
#include <hip/hip_bf16.h>

// ---------------------------------------------------------------------------
// AttnBlock2D for MI455X (gfx1250), wave32 + WMMA bf16 + async global->LDS.
//   B=8, C=512, H=W=32 (N=1024), HEADS=8, HEAD_DIM=64, INNER=512.
// Pipeline:
//   1) Q/K/V projections: per-batch GEMM  W[512x512] x X[512x1024] (+bias),
//      fp32 in -> bf16 out (workspace).
//   2) Flash-style attention per (b, head, 128-row query stripe), bf16 WMMA,
//      fp32 accumulation, online softmax; K tiles staged with
//      global_load_async_to_lds_b128 (ASYNCcnt path); row sums via a
//      P x ones(32x16) WMMA instead of lane shuffles.
//   3) Out projection: bf16 in (async-staged) -> fp32 out, + bias + residual.
// Workspace: 4 bf16 tensors of [8][512][1024]  => 32 MB required.
// ---------------------------------------------------------------------------

typedef __attribute__((ext_vector_type(16))) __bf16 v16bf;
typedef __attribute__((ext_vector_type(8)))  float  v8f;

union FragAB {                 // 16x32 A-operand or 32x16 B-operand (bf16)
    v16bf          v;
    uint4          q[2];
    unsigned short u[16];
};
union AccC {                   // 16x16 fp32 C/D tile
    v8f   v;
    float f[8];
};

__device__ __forceinline__ unsigned short f2bf(float x) {
    union { float f; unsigned int u; } c; c.f = x;
    unsigned int r = c.u + 0x7FFFu + ((c.u >> 16) & 1u);   // round-nearest-even
    return (unsigned short)(r >> 16);
}

__device__ __forceinline__ v8f wmma_bf16(const FragAB& a, const FragAB& b, v8f c) {
    // D = A(16x32) * B(32x16) + C, fp32 accumulate
    return __builtin_amdgcn_wmma_f32_16x16x32_bf16(
        /*neg_a=*/false, a.v, /*neg_b=*/false, b.v,
        /*c_mod=*/(short)0, c, /*reuse_a=*/false, /*reuse_b=*/false);
}

// Async DMA: 16 bytes global -> LDS, tracked by ASYNCcnt (no VGPR data path).
// Low 32 bits of a generic pointer to LDS are the wave-relative LDS offset
// (flat-aperture mapping), which is what the VDST operand wants.
__device__ __forceinline__ void async_load_b128(const void* g, void* l) {
    const unsigned lds = (unsigned)(unsigned long long)l;
    asm volatile("global_load_async_to_lds_b128 %0, %1, off"
                 :: "v"(lds), "v"(g)
                 : "memory");
}
__device__ __forceinline__ void wait_async0() {
    asm volatile("s_wait_asynccnt 0" ::: "memory");
}

// ---------------------------------------------------------------------------
// Stage 1 / 3: per-batch GEMM  O[b,m,n] = sum_k W[m,k] * X[b,k,n] + bias[m]
//   M=512, N=1024, K=512. Block tile 128x128, 8 waves (2x4), wave tile 64x32.
// ---------------------------------------------------------------------------
template<bool IN_BF16, bool OUT_F32>
__global__ __launch_bounds__(256)
void proj_gemm_kernel(const float* __restrict__ W,     // [512,512] fp32
                      const float* __restrict__ bias,  // [512]     fp32
                      const void*  __restrict__ Xv,    // [B,512,1024] f32|bf16
                      void*        __restrict__ Ov,    // [B,512,1024] bf16|f32
                      const float* __restrict__ resid) // [B,512,1024] or null
{
    constexpr int K = 512, LDX = 1024;
    const int b    = blockIdx.z;
    const int m0   = blockIdx.y * 128;
    const int n0   = blockIdx.x * 128;
    const int tid  = threadIdx.x;
    const int lane = tid & 31;
    const int wave = tid >> 5;
    const int wm   = wave >> 2;   // 0..1  -> 64 rows
    const int wn   = wave & 3;    // 0..3  -> 32 cols

    __shared__ __align__(16) unsigned short As[128 * 40];   // [m][k], pad 32->40
    __shared__ __align__(16) unsigned short Bs[32 * 136];   // [k][n], pad 128->136

    const size_t xbase = (size_t)b * 512 * 1024;
    const float*          Xf = (const float*)Xv + xbase;
    const unsigned short* Xb = (const unsigned short*)Xv + xbase;

    AccC acc[4][2];
    #pragma unroll
    for (int mt = 0; mt < 4; ++mt)
        #pragma unroll
        for (int nt = 0; nt < 2; ++nt)
            #pragma unroll
            for (int r = 0; r < 8; ++r) acc[mt][nt].f[r] = 0.0f;

    const int off  = (lane >= 16) ? 8 : 0;   // A-frag K-chunk select
    const int mrow = lane & 15;

    for (int kk = 0; kk < K; kk += 32) {
        __syncthreads();
        {   // B tile: X[b, kk..+31, n0..+127] -> bf16 LDS
            const int r = tid >> 3;
            const int c = (tid & 7) * 16;
            if (IN_BF16) {   // straight bf16 copy: async global->LDS DMA
                const unsigned short* src = Xb + (size_t)(kk + r) * LDX + n0 + c;
                async_load_b128(src,     &Bs[r * 136 + c]);
                async_load_b128(src + 8, &Bs[r * 136 + c + 8]);
            } else {         // fp32 -> bf16 conversion path (VGPR)
                const float* src = Xf + (size_t)(kk + r) * LDX + n0 + c;
                unsigned short tmp[16];
                #pragma unroll
                for (int j = 0; j < 16; ++j) tmp[j] = f2bf(src[j]);
                uint4* dst = (uint4*)&Bs[r * 136 + c];
                dst[0] = *(const uint4*)&tmp[0];
                dst[1] = *(const uint4*)&tmp[8];
            }
        }
        {   // A tile: W[m0..+127, kk..+31] -> bf16 LDS (fp32 convert)
            const int r = tid >> 1;
            const int c = (tid & 1) * 16;
            const float* src = W + (size_t)(m0 + r) * K + kk + c;
            unsigned short tmp[16];
            #pragma unroll
            for (int j = 0; j < 16; ++j) tmp[j] = f2bf(src[j]);
            uint4* dst = (uint4*)&As[r * 40 + c];
            dst[0] = *(const uint4*)&tmp[0];
            dst[1] = *(const uint4*)&tmp[8];
            if (kk + 32 < K)                       // global_prefetch_b8
                __builtin_prefetch(src + 32, 0, 1);
        }
        if (IN_BF16) wait_async0();                // async B tile landed in LDS
        __syncthreads();

        FragAB a[4];
        #pragma unroll
        for (int mt = 0; mt < 4; ++mt) {
            const unsigned short* p = &As[(wm * 64 + mt * 16 + mrow) * 40];
            a[mt].q[0] = *(const uint4*)&p[off];        // K = off..off+7
            a[mt].q[1] = *(const uint4*)&p[16 + off];   // K = 16+off..
        }
        #pragma unroll
        for (int nt = 0; nt < 2; ++nt) {
            FragAB bf;                                   // lane = K row
            const unsigned short* p = &Bs[wn * 32 + nt * 16 + mrow];
            #pragma unroll
            for (int s = 0; s < 16; ++s)
                bf.u[s] = p[((s < 8) ? (off + s) : (8 + off + s)) * 136];
            #pragma unroll
            for (int mt = 0; mt < 4; ++mt)
                acc[mt][nt].v = wmma_bf16(a[mt], bf, acc[mt][nt].v);
        }
    }

    // Epilogue: C layout  vgpr r, lane l -> M = r + 8*(l/16), N = l%16
    const int rhalf = (lane >> 4) * 8;
    const int ncol  = lane & 15;
    #pragma unroll
    for (int mt = 0; mt < 4; ++mt) {
        #pragma unroll
        for (int nt = 0; nt < 2; ++nt) {
            const int colg = n0 + wn * 32 + nt * 16 + ncol;
            #pragma unroll
            for (int r = 0; r < 8; ++r) {
                const int rowg = m0 + wm * 64 + mt * 16 + rhalf + r;
                float v = acc[mt][nt].f[r] + bias[rowg];
                const size_t o = xbase + (size_t)rowg * LDX + colg;
                if (OUT_F32) ((float*)Ov)[o] = v + resid[o];
                else         ((unsigned short*)Ov)[o] = f2bf(v);
            }
        }
    }
}

// ---------------------------------------------------------------------------
// Stage 2: attention. Block = (b, head, 128 q-rows), 8 waves x 16 q-rows.
// Keys looped in 32-wide tiles with online softmax; all matmuls via WMMA bf16.
// Q/K/V/AO stored as bf16 [b, head*64+d, n].
// ---------------------------------------------------------------------------
__global__ __launch_bounds__(256)
void attn_kernel(const unsigned short* __restrict__ Q,
                 const unsigned short* __restrict__ Kt,
                 const unsigned short* __restrict__ Vt,
                 unsigned short* __restrict__ AO)
{
    const int bh     = blockIdx.x;      // 0..63
    const int b      = bh >> 3;
    const int h      = bh & 7;
    const int n_base = blockIdx.y * 128;
    const int tid    = threadIdx.x;
    const int lane   = tid & 31;
    const int wave   = tid >> 5;

    __shared__ __align__(16) unsigned short Qs[128 * 80];   // [n][d], pad 64->80
    __shared__ __align__(16) unsigned short Ks[64 * 40];    // [d][m], pad 32->40
    __shared__ __align__(16) unsigned short Vs[32 * 80];    // [m][d], pad 64->80
    __shared__ __align__(16) unsigned short Ps[8 * 16 * 32];// per-wave P tile

    const size_t base = ((size_t)b * 512 + (size_t)h * 64) * 1024;
    const unsigned short* Qb = Q  + base;
    const unsigned short* Kb = Kt + base;
    const unsigned short* Vb = Vt + base;

    // Q tile [d=64][n=128] -> LDS transposed as Qs[n][d]
    #pragma unroll
    for (int i = 0; i < 4; ++i) {
        const int c  = tid * 4 + i;     // 0..1023 chunk id (8 bf16 each)
        const int d  = c >> 4;          // 0..63
        const int ns = (c & 15) * 8;    // 0..120
        uint4 v = *(const uint4*)(Qb + (size_t)d * 1024 + n_base + ns);
        const unsigned short* pv = (const unsigned short*)&v;
        #pragma unroll
        for (int j = 0; j < 8; ++j) Qs[(ns + j) * 80 + d] = pv[j];
    }
    __syncthreads();

    const int off  = (lane >= 16) ? 8 : 0;
    const int mrow = lane & 15;

    FragAB aq[2];                       // Q rows, K-dim = d (2 x 32)
    {
        const unsigned short* p = &Qs[(wave * 16 + mrow) * 80];
        aq[0].q[0] = *(const uint4*)&p[off];
        aq[0].q[1] = *(const uint4*)&p[16 + off];
        aq[1].q[0] = *(const uint4*)&p[32 + off];
        aq[1].q[1] = *(const uint4*)&p[48 + off];
    }

    FragAB ones;                        // 32x16 B operand of 1.0 (row sums)
    #pragma unroll
    for (int i = 0; i < 16; ++i) ones.u[i] = 0x3F80;

    float Mst[8];
    AccC  O[4], Lacc;
    #pragma unroll
    for (int r = 0; r < 8; ++r) { Mst[r] = -1.0e30f; Lacc.f[r] = 0.0f; }
    #pragma unroll
    for (int dt = 0; dt < 4; ++dt)
        #pragma unroll
        for (int r = 0; r < 8; ++r) O[dt].f[r] = 0.0f;

    unsigned short* Pw = &Ps[wave * 512];

    for (int m0 = 0; m0 < 1024; m0 += 32) {
        __syncthreads();
        {   // K tile -> Ks[d][m]: straight copy, async global->LDS DMA
            const int d = tid >> 2;
            const int m = (tid & 3) * 8;
            async_load_b128(Kb + (size_t)d * 1024 + m0 + m, &Ks[d * 40 + m]);
        }
        {   // V tile -> Vs[m][d] (transpose on store, VGPR path)
            const int d = tid >> 2;
            const int m = (tid & 3) * 8;
            uint4 v = *(const uint4*)(Vb + (size_t)d * 1024 + m0 + m);
            const unsigned short* pv = (const unsigned short*)&v;
            #pragma unroll
            for (int j = 0; j < 8; ++j) Vs[(m + j) * 80 + d] = pv[j];
        }
        wait_async0();                  // K tile landed in LDS
        __syncthreads();

        // S = Q * K^T : two 16x16 tiles covering 32 keys
        AccC S[2];
        #pragma unroll
        for (int t = 0; t < 2; ++t) {
            #pragma unroll
            for (int r = 0; r < 8; ++r) S[t].f[r] = 0.0f;
            FragAB bk;                                   // lane = d row
            const unsigned short* p0 = &Ks[t * 16 + mrow];
            #pragma unroll
            for (int s = 0; s < 16; ++s)
                bk.u[s] = p0[((s < 8) ? (off + s) : (8 + off + s)) * 40];
            S[t].v  = wmma_bf16(aq[0], bk, S[t].v);
            const unsigned short* p1 = &Ks[32 * 40 + t * 16 + mrow];
            #pragma unroll
            for (int s = 0; s < 16; ++s)
                bk.u[s] = p1[((s < 8) ? (off + s) : (8 + off + s)) * 40];
            S[t].v  = wmma_bf16(aq[1], bk, S[t].v);
        }

        // Online softmax; row max via xor-shuffles within each 16-lane half
        #pragma unroll
        for (int r = 0; r < 8; ++r) {
            float s0 = S[0].f[r] * 0.125f;               // 1/sqrt(64)
            float s1 = S[1].f[r] * 0.125f;
            float mx = fmaxf(s0, s1);
            #pragma unroll
            for (int sh = 1; sh < 16; sh <<= 1)
                mx = fmaxf(mx, __shfl_xor(mx, sh, 32));
            const float Mn    = fmaxf(Mst[r], mx);
            const float alpha = __expf(Mst[r] - Mn);
            Mst[r] = Mn;
            #pragma unroll
            for (int dt = 0; dt < 4; ++dt) O[dt].f[r] *= alpha;
            Lacc.f[r] *= alpha;
            const float p0 = __expf(s0 - Mn);
            const float p1 = __expf(s1 - Mn);
            const int nl = r + ((lane >> 4) << 3);       // C-layout row
            Pw[nl * 32 + (lane & 15)]      = f2bf(p0);
            Pw[nl * 32 + 16 + (lane & 15)] = f2bf(p1);
        }
        asm volatile("s_wait_dscnt 0" ::: "memory");     // P stores -> P loads

        FragAB ap;                                       // P rows, K-dim = m (32)
        {
            const unsigned short* p = &Pw[mrow * 32];
            ap.q[0] = *(const uint4*)&p[off];
            ap.q[1] = *(const uint4*)&p[16 + off];
        }
        Lacc.v = wmma_bf16(ap, ones, Lacc.v);            // row sums via WMMA
        #pragma unroll
        for (int dt = 0; dt < 4; ++dt) {                 // O += P * V
            FragAB bv;                                   // lane = m row
            const unsigned short* p = &Vs[dt * 16 + mrow];
            #pragma unroll
            for (int s = 0; s < 16; ++s)
                bv.u[s] = p[((s < 8) ? (off + s) : (8 + off + s)) * 80];
            O[dt].v = wmma_bf16(ap, bv, O[dt].v);
        }
    }

    // Normalize and store bf16 as AO[b, h*64+d, n]
    unsigned short* AOb = AO + base;
    #pragma unroll
    for (int dt = 0; dt < 4; ++dt) {
        const int d = dt * 16 + (lane & 15);
        #pragma unroll
        for (int r = 0; r < 8; ++r) {
            const int n = n_base + wave * 16 + r + ((lane >> 4) << 3);
            AOb[(size_t)d * 1024 + n] = f2bf(O[dt].f[r] / Lacc.f[r]);
        }
    }
}

// ---------------------------------------------------------------------------
extern "C" void kernel_launch(void* const* d_in, const int* in_sizes, int n_in,
                              void* d_out, int out_size, void* d_ws, size_t ws_size,
                              hipStream_t stream) {
    const float* self  = (const float*)d_in[0];
    const float* cross = (const float*)d_in[1];
    const float* Wq = (const float*)d_in[2];
    const float* bq = (const float*)d_in[3];
    const float* Wk = (const float*)d_in[4];
    const float* bk = (const float*)d_in[5];
    const float* Wv = (const float*)d_in[6];
    const float* bv = (const float*)d_in[7];
    const float* Wo = (const float*)d_in[8];
    const float* bo = (const float*)d_in[9];
    float* out = (float*)d_out;

    const size_t per = (size_t)8 * 512 * 1024;           // elements per tensor
    if (ws_size < 4 * per * sizeof(unsigned short)) return;  // need 32 MB
    unsigned short* Qw  = (unsigned short*)d_ws;
    unsigned short* Kw  = Qw + per;
    unsigned short* Vw  = Kw + per;
    unsigned short* AOw = Vw + per;

    const dim3 blk(256);
    const dim3 gproj(8, 4, 8);   // N/128, M/128, B
    const dim3 gattn(64, 8);     // B*HEADS, N/128

    proj_gemm_kernel<false, false><<<gproj, blk, 0, stream>>>(Wq, bq, self,  Qw,  nullptr);
    proj_gemm_kernel<false, false><<<gproj, blk, 0, stream>>>(Wk, bk, cross, Kw,  nullptr);
    proj_gemm_kernel<false, false><<<gproj, blk, 0, stream>>>(Wv, bv, cross, Vw,  nullptr);
    attn_kernel<<<gattn, blk, 0, stream>>>(Qw, Kw, Vw, AOw);
    proj_gemm_kernel<true, true ><<<gproj, blk, 0, stream>>>(Wo, bo, AOw, out, self);
}
